// AFMLayer_35450660061558
// MI455X (gfx1250) — matrix operand order repaired and checked by reading, not compiled, not run on hardware
//
#include <hip/hip_runtime.h>

typedef __attribute__((ext_vector_type(16))) _Float16 v16h;
typedef __attribute__((ext_vector_type(8)))  _Float16 v8h;
typedef __attribute__((ext_vector_type(8)))  float    v8f;
typedef __attribute__((ext_vector_type(4)))  float    v4f;

#define NF   40
#define DIM  64
#define ADIM 32
#define NP   780
#define PPAD 784   // 49 * 16
#define MT   49

__global__ __launch_bounds__(256) void afm_wmma_kernel(
    const float* __restrict__ in,    // [B, 40, 64]
    const float* __restrict__ W1,    // [64, 32]
    const float* __restrict__ b1,    // [32]
    const float* __restrict__ w2,    // [32]
    const float* __restrict__ pvec,  // [64]
    float* __restrict__ out)
{
    __shared__ float          s_f32[NF * DIM];
    __shared__ _Float16       s_f16[NF * DIM];
    __shared__ _Float16       s_w1t[ADIM * DIM];   // transposed: [a][d], f16
    __shared__ float          s_b1[ADIM];
    __shared__ float          s_w2[ADIM];
    __shared__ float          s_p[DIM];
    __shared__ unsigned short s_pairs[PPAD];       // i | (j << 8)
    __shared__ float          s_logits[PPAD];
    __shared__ float          s_pool[256];
    __shared__ float          s_red[64];

    const int tid  = threadIdx.x;
    const int b    = blockIdx.x;
    const int lane = tid & 31;
    const int wave = tid >> 5;
    const int g    = lane >> 4;   // lane half within wave
    const int c    = lane & 15;   // position within half

    // ---- Phase 0: stage inputs (f32 + f16), params, pair table into LDS ----
    const float* inb = in + (size_t)b * (NF * DIM);
    for (int t = tid; t < (NF * DIM) / 4; t += 256) {
        v4f v = ((const v4f*)inb)[t];
        ((v4f*)s_f32)[t] = v;
        _Float16* h = &s_f16[t * 4];
        h[0] = (_Float16)v[0]; h[1] = (_Float16)v[1];
        h[2] = (_Float16)v[2]; h[3] = (_Float16)v[3];
    }
    for (int t = tid; t < DIM * ADIM; t += 256) {
        int d = t >> 5, a = t & 31;
        s_w1t[a * DIM + d] = (_Float16)W1[t];
    }
    if (tid < ADIM) { s_b1[tid] = b1[tid]; s_w2[tid] = w2[tid]; }
    if (tid < DIM)  { s_p[tid] = pvec[tid]; }
    for (int p = tid; p < PPAD; p += 256) {
        int fi = 0, fj = 0;
        if (p < NP) {
            int rem = p, i = 0;
            while (rem >= NF - 1 - i) { rem -= NF - 1 - i; ++i; }
            fi = i; fj = i + 1 + rem;
        }
        s_pairs[p] = (unsigned short)(fi | (fj << 8));
    }
    __syncthreads();

    // ---- Phase 1: B fragments for W1 (hoisted; per wave) ----
    // B (32x16 f16): lane = column a; lane-half g holds K = g*8 + [0..7] and 16 + g*8 + [0..7].
    v16h bfrag[2][2];
    #pragma unroll
    for (int ck = 0; ck < 2; ++ck) {
        #pragma unroll
        for (int nh = 0; nh < 2; ++nh) {
            int a = nh * 16 + c;
            const v8h* col = (const v8h*)&s_w1t[a * DIM + ck * 32]; // k_local 0..31
            v8h lo = col[g];           // K = g*8 + 0..7  (local)
            v8h hi = col[2 + g];       // K = 16 + g*8 + 0..7 (local)
            v16h f;
            #pragma unroll
            for (int e = 0; e < 8; ++e) { f[e] = lo[e]; f[8 + e] = hi[e]; }
            bfrag[ck][nh] = f;
        }
    }

    // ---- Phase 2: WMMA GEMM + logits; waves own M-tiles round-robin ----
    for (int mt = wave; mt < MT; mt += 8) {
        v8f acc0 = {}; v8f acc1 = {};
        const int row = mt * 16 + c;                 // A row held by lanes c and c+16
        const unsigned pr = s_pairs[row];
        const int fi = pr & 255, fj = pr >> 8;
        const v8h* ri = (const v8h*)&s_f16[fi * DIM];
        const v8h* rj = (const v8h*)&s_f16[fj * DIM];
        #pragma unroll
        for (int ck = 0; ck < 2; ++ck) {
            // A (16x32 f16): half g holds K = g*8+[0..7] (V0-3) and 16+g*8+[0..7] (V4-7)
            v8h li = ri[ck * 4 + g],     lj = rj[ck * 4 + g];
            v8h hi = ri[ck * 4 + 2 + g], hj = rj[ck * 4 + 2 + g];
            v8h plo = li * lj;
            v8h phi = hi * hj;
            v16h af;
            #pragma unroll
            for (int e = 0; e < 8; ++e) { af[e] = plo[e]; af[8 + e] = phi[e]; }
            acc0 = __builtin_amdgcn_wmma_f32_16x16x32_f16(
                       false, af, false, bfrag[ck][0], (short)0, acc0, false, false);
            acc1 = __builtin_amdgcn_wmma_f32_16x16x32_f16(
                       false, af, false, bfrag[ck][1], (short)0, acc1, false, false);
        }
        // epilogue: h = ReLU(acc + b1), logit partial = h . w2 for this lane's column
        const float bb0 = s_b1[c],      ww0 = s_w2[c];       // nh=0 -> a = c
        const float bb1 = s_b1[16 + c], ww1 = s_w2[16 + c];  // nh=1 -> a = 16+c
        float part[8];
        #pragma unroll
        for (int r = 0; r < 8; ++r) {
            float h0 = fmaxf(acc0[r] + bb0, 0.0f);
            float h1 = fmaxf(acc1[r] + bb1, 0.0f);
            part[r] = h0 * ww0 + h1 * ww1;
        }
        // reduce across the 16 lanes of each half (masks < 16 never cross halves)
        #pragma unroll
        for (int m = 1; m < 16; m <<= 1) {
            #pragma unroll
            for (int r = 0; r < 8; ++r) part[r] += __shfl_xor(part[r], m, 32);
        }
        if (c == 0) {
            #pragma unroll
            for (int r = 0; r < 8; ++r) {
                int p = mt * 16 + g * 8 + r;   // C/D: VGPR r, half g -> M = 8g + r
                s_logits[p] = (p < NP) ? part[r] : -__builtin_inff();
            }
        }
    }
    __syncthreads();

    // ---- Phase 3: softmax over 780 pairs ----
    float lmax = -__builtin_inff();
    for (int p = tid; p < PPAD; p += 256) lmax = fmaxf(lmax, s_logits[p]);
    #pragma unroll
    for (int m = 1; m < 32; m <<= 1) lmax = fmaxf(lmax, __shfl_xor(lmax, m, 32));
    if (lane == 0) s_red[wave] = lmax;
    __syncthreads();
    float gmax = s_red[0];
    #pragma unroll
    for (int w = 1; w < 8; ++w) gmax = fmaxf(gmax, s_red[w]);
    __syncthreads();

    float lsum = 0.0f;
    for (int p = tid; p < PPAD; p += 256) {
        float e = __expf(s_logits[p] - gmax);
        s_logits[p] = e;
        lsum += e;
    }
    #pragma unroll
    for (int m = 1; m < 32; m <<= 1) lsum += __shfl_xor(lsum, m, 32);
    if (lane == 0) s_red[wave] = lsum;
    __syncthreads();
    float gsum = 0.0f;
    #pragma unroll
    for (int w = 0; w < 8; ++w) gsum += s_red[w];
    const float inv = 1.0f / gsum;
    for (int p = tid; p < PPAD; p += 256) s_logits[p] *= inv;
    __syncthreads();

    // ---- Phase 4: pooled[d] = sum_p attn[p] * x[i_p,d]*x[j_p,d]  (f32) ----
    {
        const int chunk = tid >> 6;        // 4 chunks of 196 pairs
        const int d     = tid & 63;
        const int p0 = chunk * 196;
        const int p1 = (p0 + 196 < NP) ? (p0 + 196) : NP;
        float acc = 0.0f;
        for (int p = p0; p < p1; ++p) {
            unsigned pr = s_pairs[p];
            int fi = pr & 255, fj = pr >> 8;
            acc += s_logits[p] * s_f32[fi * DIM + d] * s_f32[fj * DIM + d];
        }
        s_pool[tid] = acc;   // tid == chunk*64 + d
    }
    __syncthreads();
    if (tid < 64) {
        float pooled = s_pool[tid] + s_pool[64 + tid] + s_pool[128 + tid] + s_pool[192 + tid];
        s_red[tid] = pooled * s_p[tid];
    }
    __syncthreads();
    #pragma unroll
    for (int s = 32; s > 0; s >>= 1) {
        if (tid < s) s_red[tid] += s_red[tid + s];
        __syncthreads();
    }
    if (tid == 0) out[b] = s_red[0];
}

extern "C" void kernel_launch(void* const* d_in, const int* in_sizes, int n_in,
                              void* d_out, int out_size, void* d_ws, size_t ws_size,
                              hipStream_t stream) {
    const float* in = (const float*)d_in[0];
    const float* W1 = (const float*)d_in[1];
    const float* b1 = (const float*)d_in[2];
    const float* w2 = (const float*)d_in[3];
    const float* p  = (const float*)d_in[4];
    float* out = (float*)d_out;
    const int B = in_sizes[0] / (NF * DIM);
    afm_wmma_kernel<<<B, 256, 0, stream>>>(in, W1, b1, w2, p, out);
    (void)n_in; (void)out_size; (void)d_ws; (void)ws_size;
}
